// FrequencyGuidedAttention_28346784153715
// MI455X (gfx1250) — matrix operand order repaired and checked by reading, not compiled, not run on hardware
//
#include <hip/hip_runtime.h>
#include <hip/hip_bf16.h>
#include <stdint.h>

// ---------------- problem constants ----------------
#define BB 8
#define NN 1024
#define DD 768
#define HH 12
#define HDD 64
#define D3 2304   // 3*D

typedef __bf16 bf16_t;
typedef bf16_t v16bf __attribute__((ext_vector_type(16)));
typedef float  v8f   __attribute__((ext_vector_type(8)));

static __device__ inline unsigned short f2bf(float f) {
  unsigned int u = __float_as_uint(f);
  u += 0x7FFFu + ((u >> 16) & 1u);     // round-to-nearest-even
  return (unsigned short)(u >> 16);
}
static __device__ inline float bf2f(unsigned short h) {
  return __uint_as_float(((unsigned int)h) << 16);
}
static __device__ inline v8f zero8() {
  v8f z = {0.f,0.f,0.f,0.f,0.f,0.f,0.f,0.f};
  return z;
}

// ---- WMMA fragment loads (contiguous-K layouts used here, bf16 16x16x32) ----
// A (16x32, row-major, lda elems): lane -> M=lane&15; element e:
//   K = half*16 + e  (contiguous in K)
static __device__ inline v16bf load_frag_a(const unsigned short* A, int lda,
                                           int rowBase, int kBase, int lane) {
  int row = rowBase + (lane & 15);
  int hf  = lane >> 4;
  const unsigned short* p = A + (size_t)row * lda + kBase + hf * 16;
  union { uint4 u[2]; v16bf v; } un;
  un.u[0] = *(const uint4*)(p);
  un.u[1] = *(const uint4*)(p + 8);
  return un.v;
}
// B (32x16) given as Bt[N,K] row-major (ldb elems over K):
//   lane -> N = lane&15 ; element e -> K = (lane>>4)*16 + e  (contiguous in K)
static __device__ inline v16bf load_frag_b(const unsigned short* Bt, int ldb,
                                           int colBase, int kBase, int lane) {
  int col = colBase + (lane & 15);
  int hf  = lane >> 4;
  const unsigned short* p = Bt + (size_t)col * ldb + kBase + hf * 16;
  union { uint4 u[2]; v16bf v; } un;
  un.u[0] = *(const uint4*)(p);
  un.u[1] = *(const uint4*)(p + 8);
  return un.v;
}
static __device__ inline v8f wmma_bf16(v16bf a, v16bf b, v8f c) {
  return __builtin_amdgcn_wmma_f32_16x16x32_bf16(false, a, false, b, (short)0, c,
                                                 false, false);
}

// ---- async stage of a 64x32-elem bf16 B tile into LDS (4KB, 2 instrs/lane) --
static __device__ inline void stage_btile_async(const unsigned short* Bt, int K,
                                                int nBase, int k,
                                                unsigned short* ldsbuf, int tid) {
#pragma unroll
  for (int i = 0; i < 2; ++i) {
    int chunk = tid + 128 * i;           // 0..255 : 16-byte chunks
    int row   = chunk >> 2;              // 0..63  (B column / Bt row)
    int sub   = chunk & 3;               // 0..3   (8 bf16 each)
    const unsigned short* g = Bt + (size_t)(nBase + row) * K + k + sub * 8;
    unsigned lds = (unsigned)(uintptr_t)(ldsbuf + row * 32 + sub * 8);
    asm volatile("global_load_async_to_lds_b128 %0, %1, off"
                 :: "v"(lds), "v"((unsigned long long)(uintptr_t)g)
                 : "memory");
  }
}

// ---------------- conversion kernels ----------------
__global__ void __launch_bounds__(256)
conv_bf16(const float* __restrict__ in, unsigned short* __restrict__ out, int n) {
  int i = blockIdx.x * 256 + threadIdx.x;
  if (i < n) out[i] = f2bf(in[i]);
}

// in[K,Nc] (f32) -> out[Nc,K] (bf16), writes coalesced over K
__global__ void __launch_bounds__(256)
transpose_conv(const float* __restrict__ in, unsigned short* __restrict__ out,
               int K, int Nc) {
  int idx = blockIdx.x * 256 + threadIdx.x;
  if (idx >= K * Nc) return;
  int k = idx % K;
  int c = idx / K;
  out[idx] = f2bf(in[(size_t)k * Nc + c]);
}

// ------- WMMA GEMM: C[M,Nc] = A[M,K] @ Bt^T, 4 waves / 64x64 tile ----------
// B tile double-buffered in LDS via global_load_async_to_lds_b128; every wave
// reuses the shared tile (ds_load_b128 fragments). mode 0: bf16, 1: f32+bias.
__global__ void __launch_bounds__(128)
gemm64x64_bf16(const unsigned short* __restrict__ A,
               const unsigned short* __restrict__ Bt,
               const float* __restrict__ bias,
               void* __restrict__ Cout, int Nc, int K, int mode) {
  __shared__ __align__(16) unsigned short ldsB[2][64 * 32];
  int tid   = threadIdx.x;
  int lane  = tid & 31;
  int wid   = tid >> 5;
  int nBase = blockIdx.x * 64;
  int mBase = blockIdx.y * 64 + wid * 16;

  v8f acc0 = zero8(), acc1 = zero8(), acc2 = zero8(), acc3 = zero8();

  stage_btile_async(Bt, K, nBase, 0, &ldsB[0][0], tid);
  int buf = 0;
  for (int k = 0; k < K; k += 32) {
    if (k + 32 < K) {
      stage_btile_async(Bt, K, nBase, k + 32, &ldsB[buf ^ 1][0], tid);
      __builtin_prefetch(A + (size_t)(mBase + (lane & 15)) * K + k + 32, 0, 3);
      asm volatile("s_wait_asynccnt 0x2" ::: "memory");  // prev stage done
    } else {
      asm volatile("s_wait_asynccnt 0x0" ::: "memory");
    }
    __syncthreads();                       // tile visible to all 4 waves
    v16bf a  = load_frag_a(A, K, mBase, k, lane);
    v16bf b0 = load_frag_b(&ldsB[buf][0], 32,  0, 0, lane);
    v16bf b1 = load_frag_b(&ldsB[buf][0], 32, 16, 0, lane);
    v16bf b2 = load_frag_b(&ldsB[buf][0], 32, 32, 0, lane);
    v16bf b3 = load_frag_b(&ldsB[buf][0], 32, 48, 0, lane);
    acc0 = wmma_bf16(a, b0, acc0);
    acc1 = wmma_bf16(a, b1, acc1);
    acc2 = wmma_bf16(a, b2, acc2);
    acc3 = wmma_bf16(a, b3, acc3);
    __syncthreads();                       // everyone done before overwrite
    buf ^= 1;
  }

  int hf = lane >> 4, c0 = lane & 15;
#pragma unroll
  for (int v = 0; v < 8; ++v) {
    int row = mBase + v + 8 * hf;
    size_t ro = (size_t)row * Nc + nBase;
    if (mode == 0) {
      unsigned short* O = (unsigned short*)Cout;
      O[ro +  0 + c0] = f2bf(acc0[v]);
      O[ro + 16 + c0] = f2bf(acc1[v]);
      O[ro + 32 + c0] = f2bf(acc2[v]);
      O[ro + 48 + c0] = f2bf(acc3[v]);
    } else {
      float* O = (float*)Cout;
      O[ro +  0 + c0] = acc0[v] + bias[nBase +  0 + c0];
      O[ro + 16 + c0] = acc1[v] + bias[nBase + 16 + c0];
      O[ro + 32 + c0] = acc2[v] + bias[nBase + 32 + c0];
      O[ro + 48 + c0] = acc3[v] + bias[nBase + 48 + c0];
    }
  }
}

// ---------------- V combine: v + sigmoid(fs)*gate, stored transposed --------
__global__ void __launch_bounds__(256)
combine_v(const unsigned short* __restrict__ qkv, const float* __restrict__ g,
          const float* __restrict__ fs_p, unsigned short* __restrict__ vt) {
  int idx = blockIdx.x * 256 + threadIdx.x;
  if (idx >= BB * HH * HDD * NN) return;
  int n = idx & (NN - 1);
  int r = idx >> 10;
  int d = r & (HDD - 1);
  r >>= 6;
  int h = r % HH;
  int b = r / HH;
  float fs = 1.f / (1.f + __expf(-fs_p[0]));
  size_t rowq = (size_t)(b * NN + n);
  float v = bf2f(qkv[rowq * D3 + (2 * HH + h) * HDD + d]) +
            fs * g[rowq * DD + h * HDD + d];
  vt[idx] = f2bf(v);
}

// ---------------- fused dual-branch attention ----------------
__global__ void __launch_bounds__(32)
fga_attention(const unsigned short* __restrict__ qkv_id,
              const unsigned short* __restrict__ qkv_at,
              const unsigned short* __restrict__ vt_id,
              const unsigned short* __restrict__ vt_at,
              const float* __restrict__ temp_p,
              unsigned short* __restrict__ o_id,
              unsigned short* __restrict__ o_at) {
  int lane = threadIdx.x;
  int bid  = blockIdx.x;
  int qb = bid & 63;
  int h  = (bid >> 6) % HH;
  int b  = bid / (64 * HH);
  int nBase = qb * 16;

  const unsigned short* qid = qkv_id + (size_t)b * NN * D3 + h * HDD;
  const unsigned short* kid = qkv_id + (size_t)b * NN * D3 + (HH + h) * HDD;
  const unsigned short* qat = qkv_at + (size_t)b * NN * D3 + h * HDD;
  const unsigned short* kat = qkv_at + (size_t)b * NN * D3 + (HH + h) * HDD;
  const unsigned short* vti = vt_id + (size_t)(b * HH + h) * HDD * NN;
  const unsigned short* vta = vt_at + (size_t)(b * HH + h) * HDD * NN;

  v16bf aQi0 = load_frag_a(qid, D3, nBase,  0, lane);
  v16bf aQi1 = load_frag_a(qid, D3, nBase, 32, lane);
  v16bf aQa0 = load_frag_a(qat, D3, nBase,  0, lane);
  v16bf aQa1 = load_frag_a(qat, D3, nBase, 32, lane);

  v8f oi0 = zero8(), oi1 = zero8(), oi2 = zero8(), oi3 = zero8();
  v8f oa0 = zero8(), oa1 = zero8(), oa2 = zero8(), oa3 = zero8();
  float psI[8] = {0.f}, psA[8] = {0.f}, pd[8] = {0.f};

  __shared__ __align__(16) unsigned short ldsP[2][16 * 32];

  const float scale = 0.125f;  // 1/sqrt(64)
  int hf = lane >> 4, c0 = lane & 15;

  for (int mb = 0; mb < NN; mb += 32) {
    v8f ci0 = zero8(), ci1 = zero8(), ca0 = zero8(), ca1 = zero8();
    {
      v16bf bk;
      bk = load_frag_b(kid, D3, mb,       0, lane); ci0 = wmma_bf16(aQi0, bk, ci0);
      bk = load_frag_b(kid, D3, mb,      32, lane); ci0 = wmma_bf16(aQi1, bk, ci0);
      bk = load_frag_b(kid, D3, mb + 16,  0, lane); ci1 = wmma_bf16(aQi0, bk, ci1);
      bk = load_frag_b(kid, D3, mb + 16, 32, lane); ci1 = wmma_bf16(aQi1, bk, ci1);
      bk = load_frag_b(kat, D3, mb,       0, lane); ca0 = wmma_bf16(aQa0, bk, ca0);
      bk = load_frag_b(kat, D3, mb,      32, lane); ca0 = wmma_bf16(aQa1, bk, ca0);
      bk = load_frag_b(kat, D3, mb + 16,  0, lane); ca1 = wmma_bf16(aQa0, bk, ca1);
      bk = load_frag_b(kat, D3, mb + 16, 32, lane); ca1 = wmma_bf16(aQa1, bk, ca1);
    }
#pragma unroll
    for (int v = 0; v < 8; ++v) {
      float ei0 = __expf(fminf(fmaxf(ci0[v] * scale, -20.f), 20.f));
      float ei1 = __expf(fminf(fmaxf(ci1[v] * scale, -20.f), 20.f));
      float ea0 = __expf(fminf(fmaxf(ca0[v] * scale, -20.f), 20.f));
      float ea1 = __expf(fminf(fmaxf(ca1[v] * scale, -20.f), 20.f));
      psI[v] += ei0 + ei1;
      psA[v] += ea0 + ea1;
      pd[v]  += ei0 * ea0 + ei1 * ea1;
      int rr = (v + 8 * hf) * 32;
      ldsP[0][rr +  0 + c0] = f2bf(ei0);
      ldsP[0][rr + 16 + c0] = f2bf(ei1);
      ldsP[1][rr +  0 + c0] = f2bf(ea0);
      ldsP[1][rr + 16 + c0] = f2bf(ea1);
    }
    __syncthreads();
    v16bf aPi = load_frag_a(&ldsP[0][0], 32, 0, 0, lane);
    v16bf aPa = load_frag_a(&ldsP[1][0], 32, 0, 0, lane);
    {
      v16bf bv;
      bv = load_frag_b(vti, NN,  0, mb, lane); oi0 = wmma_bf16(aPi, bv, oi0);
      bv = load_frag_b(vti, NN, 16, mb, lane); oi1 = wmma_bf16(aPi, bv, oi1);
      bv = load_frag_b(vti, NN, 32, mb, lane); oi2 = wmma_bf16(aPi, bv, oi2);
      bv = load_frag_b(vti, NN, 48, mb, lane); oi3 = wmma_bf16(aPi, bv, oi3);
      bv = load_frag_b(vta, NN,  0, mb, lane); oa0 = wmma_bf16(aPa, bv, oa0);
      bv = load_frag_b(vta, NN, 16, mb, lane); oa1 = wmma_bf16(aPa, bv, oa1);
      bv = load_frag_b(vta, NN, 32, mb, lane); oa2 = wmma_bf16(aPa, bv, oa2);
      bv = load_frag_b(vta, NN, 48, mb, lane); oa3 = wmma_bf16(aPa, bv, oa3);
    }
    __syncthreads();
  }

#pragma unroll
  for (int v = 0; v < 8; ++v) {
    float a = psI[v], bb = psA[v], c = pd[v];
#pragma unroll
    for (int m = 1; m < 16; m <<= 1) {
      a  += __shfl_xor(a, m);
      bb += __shfl_xor(bb, m);
      c  += __shfl_xor(c, m);
    }
    psI[v] = a; psA[v] = bb; pd[v] = c;
  }

  float tclip = fminf(fmaxf(temp_p[0], 0.1f), 5.0f);
#pragma unroll
  for (int v = 0; v < 8; ++v) {
    float sI = psI[v], sA = psA[v];
    float overlap = pd[v] / (sI * sA) + 2e-8f + 1.024e-13f;
    float maskv = 1.f / (1.f + __expf(overlap * tclip));     // sigmoid(-x)
    float factor = maskv / (maskv + 1e-8f);                  // mask renorm
    float scI = factor / sI, scA = factor / sA;
    int row = nBase + v + 8 * hf;
    size_t ro = ((size_t)b * NN + row) * DD + h * HDD;
    o_id[ro +  0 + c0] = f2bf(oi0[v] * scI);
    o_id[ro + 16 + c0] = f2bf(oi1[v] * scI);
    o_id[ro + 32 + c0] = f2bf(oi2[v] * scI);
    o_id[ro + 48 + c0] = f2bf(oi3[v] * scI);
    o_at[ro +  0 + c0] = f2bf(oa0[v] * scA);
    o_at[ro + 16 + c0] = f2bf(oa1[v] * scA);
    o_at[ro + 32 + c0] = f2bf(oa2[v] * scA);
    o_at[ro + 48 + c0] = f2bf(oa3[v] * scA);
  }
}

// ---------------- LayerNorm over rows of 768 ----------------
__global__ void __launch_bounds__(256)
ln_kernel(const float* __restrict__ X, const float* __restrict__ g,
          const float* __restrict__ beta, float* __restrict__ out) {
  int row = blockIdx.x;
  const float* x = X + (size_t)row * DD;
  float* o = out + (size_t)row * DD;
  int t = threadIdx.x;
  float v0 = x[t], v1 = x[t + 256], v2 = x[t + 512];
  __shared__ float red[256];
  red[t] = v0 + v1 + v2;
  __syncthreads();
  for (int off = 128; off > 0; off >>= 1) {
    if (t < off) red[t] += red[t + off];
    __syncthreads();
  }
  float mu = red[0] * (1.f / 768.f);
  __syncthreads();
  float d0 = v0 - mu, d1 = v1 - mu, d2 = v2 - mu;
  red[t] = d0 * d0 + d1 * d1 + d2 * d2;
  __syncthreads();
  for (int off = 128; off > 0; off >>= 1) {
    if (t < off) red[t] += red[t + off];
    __syncthreads();
  }
  float inv = rsqrtf(red[0] * (1.f / 768.f) + 1e-5f);
  o[t]       = d0 * inv * g[t]       + beta[t];
  o[t + 256] = d1 * inv * g[t + 256] + beta[t + 256];
  o[t + 512] = d2 * inv * g[t + 512] + beta[t + 512];
}

// ---------------- host launcher ----------------
extern "C" void kernel_launch(void* const* d_in, const int* in_sizes, int n_in,
                              void* d_out, int out_size, void* d_ws, size_t ws_size,
                              hipStream_t stream) {
  (void)in_sizes; (void)n_in; (void)out_size; (void)ws_size;
  const float* x       = (const float*)d_in[0];
  const float* lowf    = (const float*)d_in[1];
  const float* highf   = (const float*)d_in[2];
  const float* Wqkv_id = (const float*)d_in[3];
  const float* Wqkv_at = (const float*)d_in[4];
  const float* Wfg_id  = (const float*)d_in[5];
  const float* bfg_id  = (const float*)d_in[6];
  const float* Wfg_at  = (const float*)d_in[7];
  const float* bfg_at  = (const float*)d_in[8];
  const float* Wpr_id  = (const float*)d_in[9];
  const float* bpr_id  = (const float*)d_in[10];
  const float* Wpr_at  = (const float*)d_in[11];
  const float* bpr_at  = (const float*)d_in[12];
  const float* ln_id_g = (const float*)d_in[13];
  const float* ln_id_b = (const float*)d_in[14];
  const float* ln_at_g = (const float*)d_in[15];
  const float* ln_at_b = (const float*)d_in[16];
  const float* temp_p  = (const float*)d_in[17];
  const float* fs_p    = (const float*)d_in[18];

  char* ws = (char*)d_ws;
  size_t cur = 0;
  auto alloc = [&](size_t bytes) -> void* {
    cur = (cur + 255) & ~(size_t)255;
    void* p = ws + cur;
    cur += bytes;
    return p;
  };
  const size_t N_TOK = (size_t)BB * NN;           // 8192
  const size_t SZ_XD = N_TOK * DD;                // 6,291,456

  unsigned short* xb      = (unsigned short*)alloc(SZ_XD * 2);
  unsigned short* lowb    = (unsigned short*)alloc(SZ_XD * 2);
  unsigned short* highb   = (unsigned short*)alloc(SZ_XD * 2);
  unsigned short* wqi_t   = (unsigned short*)alloc((size_t)D3 * DD * 2);
  unsigned short* wqa_t   = (unsigned short*)alloc((size_t)D3 * DD * 2);
  unsigned short* wfi_t   = (unsigned short*)alloc((size_t)DD * DD * 2);
  unsigned short* wfa_t   = (unsigned short*)alloc((size_t)DD * DD * 2);
  unsigned short* wpi_t   = (unsigned short*)alloc((size_t)DD * DD * 2);
  unsigned short* wpa_t   = (unsigned short*)alloc((size_t)DD * DD * 2);
  unsigned short* qkv_id  = (unsigned short*)alloc(N_TOK * D3 * 2);
  unsigned short* qkv_at  = (unsigned short*)alloc(N_TOK * D3 * 2);
  float*          gbuf_id = (float*)alloc(SZ_XD * 4);
  float*          gbuf_at = (float*)alloc(SZ_XD * 4);
  unsigned short* vt_id   = (unsigned short*)alloc(SZ_XD * 2);
  unsigned short* vt_at   = (unsigned short*)alloc(SZ_XD * 2);
  // lifetime aliases: xb/lowb dead after GEMMs -> attention outputs;
  // gbufs dead after combine -> projection f32 outputs.
  unsigned short* o_id   = xb;
  unsigned short* o_at   = lowb;
  float*          pr_id  = gbuf_id;
  float*          pr_at  = gbuf_at;

  const int nConv = (int)SZ_XD;
  int cb = (nConv + 255) / 256;
  conv_bf16<<<cb, 256, 0, stream>>>(x, xb, nConv);
  conv_bf16<<<cb, 256, 0, stream>>>(lowf, lowb, nConv);
  conv_bf16<<<cb, 256, 0, stream>>>(highf, highb, nConv);

  int tq = (DD * D3 + 255) / 256;
  int ts = (DD * DD + 255) / 256;
  transpose_conv<<<tq, 256, 0, stream>>>(Wqkv_id, wqi_t, DD, D3);
  transpose_conv<<<tq, 256, 0, stream>>>(Wqkv_at, wqa_t, DD, D3);
  transpose_conv<<<ts, 256, 0, stream>>>(Wfg_id, wfi_t, DD, DD);
  transpose_conv<<<ts, 256, 0, stream>>>(Wfg_at, wfa_t, DD, DD);
  transpose_conv<<<ts, 256, 0, stream>>>(Wpr_id, wpi_t, DD, DD);
  transpose_conv<<<ts, 256, 0, stream>>>(Wpr_at, wpa_t, DD, DD);

  dim3 gq(D3 / 64, (int)(N_TOK / 64));   // 36 x 128
  gemm64x64_bf16<<<gq, 128, 0, stream>>>(xb, wqi_t, nullptr, qkv_id, D3, DD, 0);
  gemm64x64_bf16<<<gq, 128, 0, stream>>>(xb, wqa_t, nullptr, qkv_at, D3, DD, 0);

  dim3 gf(DD / 64, (int)(N_TOK / 64));   // 12 x 128
  gemm64x64_bf16<<<gf, 128, 0, stream>>>(lowb,  wfi_t, bfg_id, gbuf_id, DD, DD, 1);
  gemm64x64_bf16<<<gf, 128, 0, stream>>>(highb, wfa_t, bfg_at, gbuf_at, DD, DD, 1);

  int cvb = (int)((SZ_XD + 255) / 256);
  combine_v<<<cvb, 256, 0, stream>>>(qkv_id, gbuf_id, fs_p, vt_id);
  combine_v<<<cvb, 256, 0, stream>>>(qkv_at, gbuf_at, fs_p, vt_at);

  int ablocks = BB * HH * (NN / 16);     // 6144
  fga_attention<<<ablocks, 32, 0, stream>>>(qkv_id, qkv_at, vt_id, vt_at,
                                            temp_p, o_id, o_at);

  gemm64x64_bf16<<<gf, 128, 0, stream>>>(o_id, wpi_t, bpr_id, pr_id, DD, DD, 1);
  gemm64x64_bf16<<<gf, 128, 0, stream>>>(o_at, wpa_t, bpr_at, pr_at, DD, DD, 1);

  float* out_id = (float*)d_out;
  float* out_at = out_id + SZ_XD;
  ln_kernel<<<(int)N_TOK, 256, 0, stream>>>(pr_id, ln_id_g, ln_id_b, out_id);
  ln_kernel<<<(int)N_TOK, 256, 0, stream>>>(pr_at, ln_at_g, ln_at_b, out_at);
}
